// BaselineModelWithGNN_13417477833378
// MI455X (gfx1250) — compile-verified, matrix operand order checked by reading, not run on hardware
//
#include <hip/hip_runtime.h>

typedef __attribute__((ext_vector_type(16))) __bf16 v16bf;
typedef __attribute__((ext_vector_type(8)))  __bf16 v8bf;
typedef __attribute__((ext_vector_type(8)))  float  v8f;

#define BN_EPS 1e-5f

static constexpr int kB   = 64;        // batch
static constexpr int kS   = 512;       // seq len
static constexpr int kE   = 768;       // embed
static constexpr int kD   = 384;       // gnn dim
static constexpr int kNG  = 512;       // nodes/graph
static constexpr int kN   = kB * kNG;  // 32768 nodes
static constexpr int kEdges = 1048576;
static constexpr int kK0  = 300;       // x_nodes feature dim
static constexpr int kK0P = 320;       // padded to multiple of 32

static inline int cdiv(long long a, long long b) { return (int)((a + b - 1) / b); }

// ---------------------------------------------------------------- utilities
__global__ void k_zero(float* __restrict__ p, size_t n) {
    size_t i = (size_t)blockIdx.x * blockDim.x + threadIdx.x;
    if (i < n) p[i] = 0.0f;
}

// x_nodes f32 [N,300] -> bf16 [N,320] zero-padded (removes all K-tail guards)
__global__ void k_cvt_pad(const float* __restrict__ s, __bf16* __restrict__ d) {
    int i = blockIdx.x * blockDim.x + threadIdx.x;
    if (i >= kN * kK0P) return;
    int r = i / kK0P, c = i % kK0P;
    d[i] = (c < kK0) ? (__bf16)s[(size_t)r * kK0 + c] : (__bf16)0.0f;
}

// Pack W[K,NC] f32 into WMMA B-fragment order:
//   P[((tn*KS + ks)*32 + lane)*16 + j] = W[k, tn*16 + (lane&15)]
//   k = ks*32 + (lane>=16?8:0) + (j<8 ? j : j+8), zero beyond K.
template <int K, int KP, int NC>
__global__ void k_packW(const float* __restrict__ W, __bf16* __restrict__ P) {
    constexpr int KS = KP / 32;
    int idx = blockIdx.x * blockDim.x + threadIdx.x;
    if (idx >= (NC / 16) * KS * 512) return;
    int j    = idx & 15;
    int lane = (idx >> 4) & 31;
    int ks   = (idx >> 9) % KS;
    int tn   = idx / (KS * 512);
    int k    = ks * 32 + (lane >> 4) * 8 + (j < 8 ? j : j + 8);
    int col  = tn * 16 + (lane & 15);
    P[idx] = (k < K) ? (__bf16)W[(size_t)k * NC + col] : (__bf16)0.0f;
}

// ---------------------------------------------------------------- sentence branch
__global__ void k_sentence_mean(const float* __restrict__ last,
                                const float* __restrict__ first,
                                float* __restrict__ h) {
    int i = blockIdx.x * blockDim.x + threadIdx.x;
    if (i >= kB * kE) return;
    int b = i / kE, e = i % kE;
    const float* lp = last  + (size_t)b * kS * kE + e;
    const float* fp = first + (size_t)b * kS * kE + e;
    float s = 0.0f;
    for (int t = 0; t < kS; ++t) s += lp[(size_t)t * kE] + fp[(size_t)t * kE];
    h[i] = s * (0.5f / (float)kS);
}

// ---------------------------------------------------------------- WMMA GEMM
// C[M,NC](f32) = (relu?)(A[M,KP](bf16) @ Wpacked + bias)
// wave computes a 16x64 strip (4 n-tiles); 1-deep software pipeline:
// loads for k-step ks+1 are issued before the wmmas of ks, so the compiler
// can use partial s_wait_loadcnt instead of waiting to zero before each wmma.
template <int KP, int NC, bool RELU, bool BIAS>
__global__ __launch_bounds__(256) void k_gemm(const __bf16* __restrict__ A,
                                              const __bf16* __restrict__ Wp,
                                              const float* __restrict__ bias,
                                              float* __restrict__ C, int M) {
    constexpr int KS = KP / 32;
    constexpr int NT = 4;                    // n-tiles per wave
    constexpr int NGRP = (NC / 16) / NT;     // wave columns
    int wid = blockIdx.x * 8 + (threadIdx.x >> 5);
    int total = (M / 16) * NGRP;
    if (wid >= total) return;                // wave-uniform
    int lane = threadIdx.x & 31;
    int half = lane >> 4;                    // k sub-block select
    int l15  = lane & 15;

    int tm = wid / NGRP, tg = wid % NGRP;
    const __bf16* ap = A + (size_t)(tm * 16 + l15) * KP + half * 8;
    const __bf16* wp = Wp + ((size_t)(tg * NT) * KS * 32 + lane) * 16;

    v8f acc[NT];
#pragma unroll
    for (int t = 0; t < NT; ++t) acc[t] = (v8f){0.f,0.f,0.f,0.f,0.f,0.f,0.f,0.f};

    auto loadA = [&](int ks) {
        v8bf lo = *(const v8bf*)(ap + ks * 32);
        v8bf hi = *(const v8bf*)(ap + ks * 32 + 16);
        v16bf f;
#pragma unroll
        for (int j = 0; j < 8; ++j) { f[j] = lo[j]; f[j + 8] = hi[j]; }
        return f;
    };

    // prologue: fragments for ks = 0
    v16bf afc = loadA(0);
    v16bf bfc[NT];
#pragma unroll
    for (int t = 0; t < NT; ++t)
        bfc[t] = *(const v16bf*)(wp + (size_t)t * KS * 512);

#pragma unroll 2
    for (int ks = 0; ks < KS - 1; ++ks) {
        // issue next k-step's loads first (in-order completion -> partial waits)
        v16bf afn = loadA(ks + 1);
        v16bf bfn[NT];
#pragma unroll
        for (int t = 0; t < NT; ++t)
            bfn[t] = *(const v16bf*)(wp + ((size_t)t * KS + ks + 1) * 512);
        __builtin_prefetch(ap + (ks + 2) * 32, 0, 1);   // global_prefetch A stream
#pragma unroll
        for (int t = 0; t < NT; ++t)
            acc[t] = __builtin_amdgcn_wmma_f32_16x16x32_bf16(
                false, afc, false, bfc[t], (short)0, acc[t], false, false);
        afc = afn;
#pragma unroll
        for (int t = 0; t < NT; ++t) bfc[t] = bfn[t];
    }
#pragma unroll
    for (int t = 0; t < NT; ++t)
        acc[t] = __builtin_amdgcn_wmma_f32_16x16x32_bf16(
            false, afc, false, bfc[t], (short)0, acc[t], false, false);

    int rbase = tm * 16 + half * 8;
#pragma unroll
    for (int t = 0; t < NT; ++t) {
        int bcol = (tg * NT + t) * 16 + l15;
        float bs = BIAS ? bias[bcol] : 0.0f;
#pragma unroll
        for (int r = 0; r < 8; ++r) {
            float v = acc[t][r] + bs;
            if (RELU) v = fmaxf(v, 0.0f);
            C[(size_t)(rbase + r) * NC + bcol] = v;
        }
    }
}

// ---------------------------------------------------------------- batchnorm (training stats)
__global__ void k_colstats(const float* __restrict__ X, float* __restrict__ stats,
                           int M, int NC) {
    int col = threadIdx.x;
    int r0 = blockIdx.x * 64;
    int r1 = r0 + 64; if (r1 > M) r1 = M;
    float s = 0.0f, s2 = 0.0f;
    for (int r = r0; r < r1; ++r) {
        float v = X[(size_t)r * NC + col];
        s += v; s2 += v * v;
    }
    atomicAdd(&stats[col], s);
    atomicAdd(&stats[NC + col], s2);
}

// BN -> bf16 output (post-BN f32 is never needed: it only feeds the next GEMM/gather)
__global__ void k_bn_bf16(const float* __restrict__ X, const float* __restrict__ stats,
                          const float* __restrict__ g, const float* __restrict__ b,
                          __bf16* __restrict__ out, int M, int NC) {
    size_t i = (size_t)blockIdx.x * blockDim.x + threadIdx.x;
    if (i >= (size_t)M * NC) return;
    int col = (int)(i % NC);
    float inv = 1.0f / (float)M;
    float m = stats[col] * inv;
    float v = stats[NC + col] * inv - m * m;
    out[i] = (__bf16)(g[col] * (X[i] - m) * rsqrtf(v + BN_EPS) + b[col]);
}

// BN in place, f32 (for outc, which is consumed by an f32 add)
__global__ void k_bn_apply(float* __restrict__ X, const float* __restrict__ stats,
                           const float* __restrict__ g, const float* __restrict__ b,
                           int M, int NC) {
    size_t i = (size_t)blockIdx.x * blockDim.x + threadIdx.x;
    if (i >= (size_t)M * NC) return;
    int col = (int)(i % NC);
    float inv = 1.0f / (float)M;
    float m = stats[col] * inv;
    float v = stats[NC + col] * inv - m * m;
    X[i] = g[col] * (X[i] - m) * rsqrtf(v + BN_EPS) + b[col];
}

// att = BN(hsent) + outc
__global__ void k_bn_add(const float* __restrict__ hs, const float* __restrict__ stats,
                         const float* __restrict__ g, const float* __restrict__ b,
                         const float* __restrict__ outc, float* __restrict__ att,
                         int M, int NC) {
    size_t i = (size_t)blockIdx.x * blockDim.x + threadIdx.x;
    if (i >= (size_t)M * NC) return;
    int col = (int)(i % NC);
    float inv = 1.0f / (float)M;
    float m = stats[col] * inv;
    float v = stats[NC + col] * inv - m * m;
    att[i] = g[col] * (hs[i] - m) * rsqrtf(v + BN_EPS) + b[col] + outc[i];
}

// ---------------------------------------------------------------- graph degree / scatter
__global__ void k_degcount(const int* __restrict__ dst, float* __restrict__ deg) {
    int e = blockIdx.x * blockDim.x + threadIdx.x;
    if (e < kEdges) atomicAdd(&deg[dst[e]], 1.0f);
}

__global__ void k_degfin(float* __restrict__ deg) {
    int i = blockIdx.x * blockDim.x + threadIdx.x;
    if (i < kN) deg[i] = rsqrtf(deg[i] + 1.0f);
}

__global__ void k_scatter(const float* __restrict__ y, const int* __restrict__ src,
                          const int* __restrict__ dst, const float* __restrict__ dis,
                          float* __restrict__ agg) {
    int e = blockIdx.x;
    int s = src[e], d = dst[e];
    float norm = dis[s] * dis[d];
    const float* yp = y + (size_t)s * kD;
    float* op = agg + (size_t)d * kD;
    for (int f = threadIdx.x; f < kD; f += blockDim.x)
        atomicAdd(&op[f], yp[f] * norm);
}

__global__ void k_selfloop(float* __restrict__ agg, const float* __restrict__ y,
                           const float* __restrict__ dis, const float* __restrict__ bias) {
    size_t i = (size_t)blockIdx.x * blockDim.x + threadIdx.x;
    if (i >= (size_t)kN * kD) return;
    int node = (int)(i / kD), f = (int)(i % kD);
    float di = dis[node];
    float v = agg[i] + y[i] * di * di + bias[f];
    agg[i] = fmaxf(v, 0.0f);
}

// ---------------------------------------------------------------- masked-node gather (bf16)
__global__ void k_gather(const __bf16* __restrict__ X, const int* __restrict__ mask,
                         __bf16* __restrict__ flat) {
    __shared__ int sel[2];
    int b = blockIdx.x;
    if (threadIdx.x == 0) {
        int c = 0;
        for (int j = 0; j < kNG && c < 2; ++j)
            if (mask[b * kNG + j] != 0) sel[c++] = j;
        if (c == 0) { sel[0] = 0; sel[1] = 0; }
        else if (c == 1) sel[1] = sel[0];
    }
    __syncthreads();
    for (int f = threadIdx.x; f < kE; f += blockDim.x) {
        int which = f / kD;
        int node = b * kNG + sel[which];
        flat[(size_t)b * kE + f] = X[(size_t)node * kD + (f - which * kD)];
    }
}

// ---------------------------------------------------------------- classifier head
__global__ void k_final(const float* __restrict__ att, const float* __restrict__ w,
                        const float* __restrict__ bo, float* __restrict__ out) {
    int i = blockIdx.x * blockDim.x + threadIdx.x;
    if (i >= kB * 3) return;
    int b = i / 3, c = i % 3;
    float s = bo[c];
    const float* ap = att + (size_t)b * kE;
    for (int e = 0; e < kE; ++e) s += ap[e] * w[e * 3 + c];
    out[i] = s;
}

// ================================================================ launch
extern "C" void kernel_launch(void* const* d_in, const int* in_sizes, int n_in,
                              void* d_out, int out_size, void* d_ws, size_t ws_size,
                              hipStream_t stream) {
    const float* last_h   = (const float*)d_in[0];
    const float* first_h  = (const float*)d_in[1];
    const float* x_nodes  = (const float*)d_in[2];
    const int*   edge_idx = (const int*)d_in[3];
    const int*   mask     = (const int*)d_in[4];
    const float* w_pre1   = (const float*)d_in[5];
    const float* b_pre1   = (const float*)d_in[6];
    const float* w_pre2   = (const float*)d_in[7];
    const float* b_pre2   = (const float*)d_in[8];
    const float* w_conv   = (const float*)d_in[9];
    const float* b_conv   = (const float*)d_in[10];
    const float* bng_g    = (const float*)d_in[11];
    const float* bng_b    = (const float*)d_in[12];
    const float* w_post1  = (const float*)d_in[13];
    const float* b_post1  = (const float*)d_in[14];
    const float* w_post2  = (const float*)d_in[15];
    const float* b_post2  = (const float*)d_in[16];
    const float* w_cat    = (const float*)d_in[17];
    const float* b_cat    = (const float*)d_in[18];
    const float* bn_g     = (const float*)d_in[19];
    const float* bn_b     = (const float*)d_in[20];
    const float* w_out    = (const float*)d_in[21];
    const float* b_out    = (const float*)d_in[22];
    float* out = (float*)d_out;

    const int* srcp = edge_idx;
    const int* dstp = edge_idx + kEdges;

    // ---- workspace layout (f32 region, then 32B-aligned bf16 region)
    const size_t ND = (size_t)kN * kD;
    float* ws    = (float*)d_ws;
    float* F1    = ws;                  // [N, D] gemm out / y
    float* F2    = F1 + ND;             // [N, D] scatter accumulator
    float* dis   = F2 + ND;             // [N]
    float* stats = dis + kN;            // [2*E]
    float* hs    = stats + 2 * kE;      // [B, E]
    float* outc  = hs + kB * kE;        // [B, E]
    float* att   = outc + kB * kE;      // [B, E]
    uintptr_t pa = ((uintptr_t)(att + kB * kE) + 31) & ~(uintptr_t)31;
    __bf16* Xpad   = (__bf16*)pa;                      // [N, 320]
    __bf16* Xbf    = Xpad + (size_t)kN * kK0P;         // [N, 384]
    __bf16* flatbf = Xbf + ND;                         // [B, 768]
    __bf16* wp_pre1  = flatbf + kB * kE;               // 24*10*512
    __bf16* wp_pre2  = wp_pre1 + (kD/16) * (kK0P/32) * 512;
    __bf16* wp_conv  = wp_pre2 + (kD/16) * (kD/32) * 512;
    __bf16* wp_post1 = wp_conv + (kD/16) * (kD/32) * 512;
    __bf16* wp_post2 = wp_post1 + (kD/16) * (kD/32) * 512;
    __bf16* wp_cat   = wp_post2 + (kD/16) * (kD/32) * 512;

    // ---- weight packing into WMMA fragment order (bf16, zero-padded K)
    k_packW<kK0, kK0P, kD><<<cdiv((kD/16)*(kK0P/32)*512, 256), 256, 0, stream>>>(w_pre1, wp_pre1);
    k_packW<kD, kD, kD><<<cdiv((kD/16)*(kD/32)*512, 256), 256, 0, stream>>>(w_pre2, wp_pre2);
    k_packW<kD, kD, kD><<<cdiv((kD/16)*(kD/32)*512, 256), 256, 0, stream>>>(w_conv + 2 * kD * kD, wp_conv);
    k_packW<kD, kD, kD><<<cdiv((kD/16)*(kD/32)*512, 256), 256, 0, stream>>>(w_post1, wp_post1);
    k_packW<kD, kD, kD><<<cdiv((kD/16)*(kD/32)*512, 256), 256, 0, stream>>>(w_post2, wp_post2);
    k_packW<kE, kE, kE><<<cdiv((kE/16)*(kE/32)*512, 256), 256, 0, stream>>>(w_cat, wp_cat);

    // ---- activation prep + sentence branch
    k_cvt_pad<<<cdiv((long long)kN * kK0P, 256), 256, 0, stream>>>(x_nodes, Xpad);
    k_sentence_mean<<<cdiv(kB * kE, 256), 256, 0, stream>>>(last_h, first_h, hs);

    const int gB = cdiv((kN / 16) * ((kD / 16) / 4), 8);   // big-GEMM grid (4 n-tiles/wave)

    // ---- pre1: F1 = relu(Xpad @ W + b); Xbf = BN(F1) (bng row 0)
    k_gemm<kK0P, kD, true, true><<<gB, 256, 0, stream>>>(Xpad, wp_pre1, b_pre1, F1, kN);
    k_zero<<<cdiv(2 * kE, 256), 256, 0, stream>>>(stats, 2 * kE);
    k_colstats<<<cdiv(kN, 64), kD, 0, stream>>>(F1, stats, kN, kD);
    k_bn_bf16<<<cdiv((long long)ND, 256), 256, 0, stream>>>(F1, stats, bng_g, bng_b, Xbf, kN, kD);

    // ---- pre2 (bng row 1); Xbf = x for the conv
    k_gemm<kD, kD, true, true><<<gB, 256, 0, stream>>>(Xbf, wp_pre2, b_pre2, F1, kN);
    k_zero<<<cdiv(2 * kE, 256), 256, 0, stream>>>(stats, 2 * kE);
    k_colstats<<<cdiv(kN, 64), kD, 0, stream>>>(F1, stats, kN, kD);
    k_bn_bf16<<<cdiv((long long)ND, 256), 256, 0, stream>>>(F1, stats, bng_g + kD, bng_b + kD, Xbf, kN, kD);

    // ---- degree norm of (A+I)
    k_zero<<<cdiv(kN, 256), 256, 0, stream>>>(dis, kN);
    k_degcount<<<cdiv(kEdges, 256), 256, 0, stream>>>(dstp, dis);
    k_degfin<<<cdiv(kN, 256), 256, 0, stream>>>(dis);

    // ---- GCN conv (only i=2 is live; i=0,1 dead in reference) -> bng row 4
    k_gemm<kD, kD, false, false><<<gB, 256, 0, stream>>>(Xbf, wp_conv, nullptr, F1, kN);
    k_zero<<<cdiv((long long)ND, 256), 256, 0, stream>>>(F2, ND);
    k_scatter<<<kEdges, 128, 0, stream>>>(F1, srcp, dstp, dis, F2);
    k_selfloop<<<cdiv((long long)ND, 256), 256, 0, stream>>>(F2, F1, dis, b_conv + 2 * kD);
    k_zero<<<cdiv(2 * kE, 256), 256, 0, stream>>>(stats, 2 * kE);
    k_colstats<<<cdiv(kN, 64), kD, 0, stream>>>(F2, stats, kN, kD);
    k_bn_bf16<<<cdiv((long long)ND, 256), 256, 0, stream>>>(F2, stats, bng_g + 4 * kD, bng_b + 4 * kD, Xbf, kN, kD);

    // ---- post1 (bng row 5)
    k_gemm<kD, kD, true, true><<<gB, 256, 0, stream>>>(Xbf, wp_post1, b_post1, F1, kN);
    k_zero<<<cdiv(2 * kE, 256), 256, 0, stream>>>(stats, 2 * kE);
    k_colstats<<<cdiv(kN, 64), kD, 0, stream>>>(F1, stats, kN, kD);
    k_bn_bf16<<<cdiv((long long)ND, 256), 256, 0, stream>>>(F1, stats, bng_g + 5 * kD, bng_b + 5 * kD, Xbf, kN, kD);

    // ---- post2 (bng row 6); final node features bf16 in Xbf
    k_gemm<kD, kD, true, true><<<gB, 256, 0, stream>>>(Xbf, wp_post2, b_post2, F1, kN);
    k_zero<<<cdiv(2 * kE, 256), 256, 0, stream>>>(stats, 2 * kE);
    k_colstats<<<cdiv(kN, 64), kD, 0, stream>>>(F1, stats, kN, kD);
    k_bn_bf16<<<cdiv((long long)ND, 256), 256, 0, stream>>>(F1, stats, bng_g + 6 * kD, bng_b + 6 * kD, Xbf, kN, kD);

    // ---- gather 2 masked nodes per graph -> flat [B, E] (bf16)
    k_gather<<<kB, 256, 0, stream>>>(Xbf, mask, flatbf);

    // ---- outc = BN(relu(flat @ w_cat + b_cat)) (bn row 0), stays f32
    k_gemm<kE, kE, true, true><<<cdiv((kB / 16) * ((kE / 16) / 4), 8), 256, 0, stream>>>(flatbf, wp_cat, b_cat, outc, kB);
    k_zero<<<cdiv(2 * kE, 256), 256, 0, stream>>>(stats, 2 * kE);
    k_colstats<<<cdiv(kB, 64), kE, 0, stream>>>(outc, stats, kB, kE);
    k_bn_apply<<<cdiv(kB * kE, 256), 256, 0, stream>>>(outc, stats, bn_g, bn_b, kB, kE);

    // ---- att = BN(H_sent) (bn row 1) + outc
    k_zero<<<cdiv(2 * kE, 256), 256, 0, stream>>>(stats, 2 * kE);
    k_colstats<<<cdiv(kB, 64), kE, 0, stream>>>(hs, stats, kB, kE);
    k_bn_add<<<cdiv(kB * kE, 256), 256, 0, stream>>>(hs, stats, bn_g + kE, bn_b + kE, outc, att, kB, kE);

    // ---- logits
    k_final<<<1, 192, 0, stream>>>(att, w_out, b_out, out);

    (void)in_sizes; (void)n_in; (void)out_size; (void)ws_size;
}
